// PositionAwareSelfAttention_80109730005018
// MI455X (gfx1250) — matrix operand order repaired and checked by reading, not compiled, not run on hardware
//
#include <hip/hip_runtime.h>
#include <hip/hip_bf16.h>

#define BATCH 4
#define SEQ   2048
#define CH    512
#define HEADS 8
#define HDIM  64
#define ROWS  (BATCH * SEQ)   // 8192

typedef __attribute__((ext_vector_type(16))) _Float16 v16h;
typedef __attribute__((ext_vector_type(8)))  _Float16 h8;
typedef __attribute__((ext_vector_type(8)))  float    v8f;

__device__ __forceinline__ v8f vzero8() {
    v8f z;
#pragma unroll
    for (int i = 0; i < 8; ++i) z[i] = 0.0f;
    return z;
}

__device__ __forceinline__ v16h combine16(h8 lo, h8 hi) {
    v16h r;
#pragma unroll
    for (int i = 0; i < 8; ++i) { r[i] = lo[i]; r[i + 8] = hi[i]; }
    return r;
}

// A-fragment (16x32 f16): lane L holds row (L&15); elements 0-15 = K kh..kh+15,
// kh = (L>=16)*16.  base -> tile row0,k0.
__device__ __forceinline__ v16h load_a_frag(const _Float16* __restrict__ base, int ld) {
    int lane = threadIdx.x & 31;
    int row  = lane & 15;
    int kh   = (lane >> 4) * 16;
    const _Float16* p = base + row * ld + kh;
    h8 lo = *(const h8*)(p);
    h8 hi = *(const h8*)(p + 8);
    return combine16(lo, hi);
}

// B-fragment (32x16 f16) from an NT (row-major-over-K) matrix: lane L holds
// column N=(L&15); element e = K ((L>=16)*16 + e).  base -> row n0, col k0.
__device__ __forceinline__ v16h load_b_frag(const _Float16* __restrict__ base, int ld) {
    int lane = threadIdx.x & 31;
    int n    = lane & 15;
    int kk   = (lane >> 4) * 16;
    return *(const v16h*)(base + n * ld + kk);
}

__device__ __forceinline__ v8f wmma32(v16h a, v16h b, v8f c) {
    return __builtin_amdgcn_wmma_f32_16x16x32_f16(false, a, false, b, (short)0, c, false, false);
}

// ---------------------------------------------------------------- fp32 -> f16
__global__ void cvt_f16_kernel(const float* __restrict__ src,
                               _Float16* __restrict__ dst, int n) {
    int i = blockIdx.x * blockDim.x + threadIdx.x;
    int stride = gridDim.x * blockDim.x;
    for (; i < n; i += stride) dst[i] = (_Float16)src[i];
}

// ------------------------------------------------ QKV projection GEMM (NT)
// out[m,o] = sum_k X[m,k] * W[o,k] + bias[o]
// 4 waves / block, each wave computes a 64x64 tile (4x4 WMMA register block):
// 16 WMMAs per 16 b128 loads per k-step -> 4x A/B reuse.
// mode 0: write [b][h][n][d] f16 ; mode 1: write [b][h][d][n] f16 (V transposed)
__global__ __launch_bounds__(128)
void gemm_qkv_kernel(const _Float16* __restrict__ X, const _Float16* __restrict__ W,
                     const float* __restrict__ bias, _Float16* __restrict__ out, int mode) {
    const int K = CH, ldx = CH, ldw = CH;
    int wave = threadIdx.x >> 5;
    int lane = threadIdx.x & 31;
    int rbase = blockIdx.x * 128 + (wave & 1) * 64;
    int cbase = blockIdx.y * 128 + (wave >> 1) * 64;

    v8f acc[4][4];
#pragma unroll
    for (int i = 0; i < 4; ++i)
#pragma unroll
        for (int j = 0; j < 4; ++j) acc[i][j] = vzero8();

    for (int k0 = 0; k0 < K; k0 += 32) {
        v16h a[4], bf[4];
#pragma unroll
        for (int i = 0; i < 4; ++i)
            a[i] = load_a_frag(X + (size_t)(rbase + 16 * i) * ldx + k0, ldx);
#pragma unroll
        for (int j = 0; j < 4; ++j)
            bf[j] = load_b_frag(W + (size_t)(cbase + 16 * j) * ldw + k0, ldw);
#pragma unroll
        for (int i = 0; i < 4; ++i)
#pragma unroll
            for (int j = 0; j < 4; ++j)
                acc[i][j] = wmma32(a[i], bf[j], acc[i][j]);
    }

    int nloc = lane & 15, hs = lane >> 4;
#pragma unroll
    for (int rt = 0; rt < 4; ++rt)
#pragma unroll
        for (int ct = 0; ct < 4; ++ct)
#pragma unroll
            for (int r = 0; r < 8; ++r) {
                int row = rbase + rt * 16 + r + 8 * hs;
                int col = cbase + ct * 16 + nloc;
                float v = acc[rt][ct][r] + bias[col];
                int b = row >> 11, n = row & (SEQ - 1);
                int h = col >> 6,  d = col & (HDIM - 1);
                size_t idx;
                if (mode == 0) idx = ((size_t)(b * HEADS + h) * SEQ + n) * HDIM + d;
                else           idx = ((size_t)(b * HEADS + h) * HDIM + d) * SEQ + n;
                out[idx] = (_Float16)v;
            }
}

// --------------------------------------------- flash attention, one wave / 16 q rows
__global__ __launch_bounds__(128)
void flash_attn_kernel(const _Float16* __restrict__ Qh, const _Float16* __restrict__ Kh,
                       const _Float16* __restrict__ Vt, const float* __restrict__ qp,
                       _Float16* __restrict__ Oh) {
    __shared__ __align__(16) _Float16 Plds[4][16][72];   // per-wave 16x64 P tile (+pad)

    int wave = threadIdx.x >> 5;
    int lane = threadIdx.x & 31;
    int nloc = lane & 15, hs = lane >> 4;
    int bh = blockIdx.y;
    int b = bh >> 3, h = bh & 7;
    int qbase = blockIdx.x * 64 + wave * 16;

    const _Float16* Qb = Qh + (size_t)bh * SEQ * HDIM;
    const _Float16* Kb = Kh + (size_t)bh * SEQ * HDIM;
    const _Float16* Vb = Vt + (size_t)bh * HDIM * SEQ;

    v16h qf0 = load_a_frag(Qb + (size_t)qbase * HDIM +  0, HDIM);
    v16h qf1 = load_a_frag(Qb + (size_t)qbase * HDIM + 32, HDIM);

    float qx[8], qy[8], m8[8], l8[8];
    v8f O[4];
#pragma unroll
    for (int r = 0; r < 8; ++r) {
        int g = qbase + r + 8 * hs;
        qx[r] = qp[((size_t)(b * SEQ + g)) * 4 + 0];
        qy[r] = qp[((size_t)(b * SEQ + g)) * 4 + 1];
        m8[r] = -1e30f;
        l8[r] = 0.0f;
    }
#pragma unroll
    for (int dt = 0; dt < 4; ++dt) O[dt] = vzero8();

    for (int kb = 0; kb < SEQ; kb += 64) {
        // ---- S = (Q K^T)/8 + bias, tiles of 16x16
        v8f S[4];
#pragma unroll
        for (int ct = 0; ct < 4; ++ct) {
            v16h bk0 = load_b_frag(Kb + (size_t)(kb + ct * 16) * HDIM +  0, HDIM);
            v16h bk1 = load_b_frag(Kb + (size_t)(kb + ct * 16) * HDIM + 32, HDIM);
            v8f z = vzero8();
            z = wmma32(qf0, bk0, z);
            S[ct] = wmma32(qf1, bk1, z);
        }

        float p[4][8], rm[8];
#pragma unroll
        for (int r = 0; r < 8; ++r) rm[r] = -1e30f;
#pragma unroll
        for (int ct = 0; ct < 4; ++ct) {
            int key = kb + ct * 16 + nloc;
            float kx = qp[((size_t)(b * SEQ + key)) * 4 + 0];
            float ky = qp[((size_t)(b * SEQ + key)) * 4 + 1];
#pragma unroll
            for (int r = 0; r < 8; ++r) {
                float dx = fabsf(qx[r] - kx), dy = fabsf(qy[r] - ky);
                float biasv = -0.1f * sqrtf(dx * dx + dy * dy + 1e-8f);
                float s = S[ct][r] * 0.125f + biasv;   // 1/sqrt(64)
                p[ct][r] = s;
                rm[r] = fmaxf(rm[r], s);
            }
        }
        // row max across 16 lanes (row lives in one half-wave)
#pragma unroll
        for (int r = 0; r < 8; ++r) {
            float v = rm[r];
            v = fmaxf(v, __shfl_xor(v, 1, 32));
            v = fmaxf(v, __shfl_xor(v, 2, 32));
            v = fmaxf(v, __shfl_xor(v, 4, 32));
            v = fmaxf(v, __shfl_xor(v, 8, 32));
            rm[r] = v;
        }
        float alpha[8], rs[8];
#pragma unroll
        for (int r = 0; r < 8; ++r) {
            float mn = fmaxf(m8[r], rm[r]);
            alpha[r] = __expf(m8[r] - mn);
            m8[r] = mn;
            rs[r] = 0.0f;
        }
#pragma unroll
        for (int ct = 0; ct < 4; ++ct)
#pragma unroll
            for (int r = 0; r < 8; ++r) {
                float e = __expf(p[ct][r] - m8[r]);
                p[ct][r] = e;
                rs[r] += e;
            }
#pragma unroll
        for (int r = 0; r < 8; ++r) {
            float v = rs[r];
            v += __shfl_xor(v, 1, 32);
            v += __shfl_xor(v, 2, 32);
            v += __shfl_xor(v, 4, 32);
            v += __shfl_xor(v, 8, 32);
            l8[r] = l8[r] * alpha[r] + v;
        }
#pragma unroll
        for (int dt = 0; dt < 4; ++dt)
#pragma unroll
            for (int r = 0; r < 8; ++r) O[dt][r] *= alpha[r];

        // ---- P (C-layout) -> LDS -> A-fragment layout
#pragma unroll
        for (int ct = 0; ct < 4; ++ct)
#pragma unroll
            for (int r = 0; r < 8; ++r)
                Plds[wave][r + 8 * hs][ct * 16 + nloc] = (_Float16)p[ct][r];
        __builtin_amdgcn_wave_barrier();   // wave-internal DS ordering fence

        const _Float16* pr = &Plds[wave][nloc][0];
        int kh = hs * 16;
        v16h pf0 = combine16(*(const h8*)(pr + kh),       *(const h8*)(pr + kh + 8));
        v16h pf1 = combine16(*(const h8*)(pr + 32 + kh),  *(const h8*)(pr + 32 + kh + 8));

        // ---- O += P @ V  (B-fragment from V^T rows, contiguous)
#pragma unroll
        for (int dt = 0; dt < 4; ++dt) {
            v16h v0 = *(const v16h*)(Vb + (size_t)(dt * 16 + nloc) * SEQ + kb +  0 + hs * 16);
            v16h v1 = *(const v16h*)(Vb + (size_t)(dt * 16 + nloc) * SEQ + kb + 32 + hs * 16);
            O[dt] = wmma32(pf0, v0, O[dt]);
            O[dt] = wmma32(pf1, v1, O[dt]);
        }
    }

    // ---- normalize + merge heads into [b][n][C] f16
#pragma unroll
    for (int dt = 0; dt < 4; ++dt)
#pragma unroll
        for (int r = 0; r < 8; ++r) {
            int g = qbase + r + 8 * hs;
            int d = dt * 16 + nloc;
            float v = O[dt][r] / l8[r];
            Oh[((size_t)(b * SEQ + g)) * CH + h * HDIM + d] = (_Float16)v;
        }
}

// --------------------------------- output projection + bias + residual (fp32 H)
// Same 4x4 register blocking as gemm_qkv_kernel.
__global__ __launch_bounds__(128)
void gemm_out_kernel(const _Float16* __restrict__ A, const _Float16* __restrict__ W,
                     const float* __restrict__ bias, const float* __restrict__ resid,
                     float* __restrict__ H) {
    const int K = CH, lda = CH, ldw = CH;
    int wave = threadIdx.x >> 5;
    int lane = threadIdx.x & 31;
    int rbase = blockIdx.x * 128 + (wave & 1) * 64;
    int cbase = blockIdx.y * 128 + (wave >> 1) * 64;

    v8f acc[4][4];
#pragma unroll
    for (int i = 0; i < 4; ++i)
#pragma unroll
        for (int j = 0; j < 4; ++j) acc[i][j] = vzero8();

    for (int k0 = 0; k0 < K; k0 += 32) {
        v16h a[4], bf[4];
#pragma unroll
        for (int i = 0; i < 4; ++i)
            a[i] = load_a_frag(A + (size_t)(rbase + 16 * i) * lda + k0, lda);
#pragma unroll
        for (int j = 0; j < 4; ++j)
            bf[j] = load_b_frag(W + (size_t)(cbase + 16 * j) * ldw + k0, ldw);
#pragma unroll
        for (int i = 0; i < 4; ++i)
#pragma unroll
            for (int j = 0; j < 4; ++j)
                acc[i][j] = wmma32(a[i], bf[j], acc[i][j]);
    }

    int nloc = lane & 15, hs = lane >> 4;
#pragma unroll
    for (int rt = 0; rt < 4; ++rt)
#pragma unroll
        for (int ct = 0; ct < 4; ++ct)
#pragma unroll
            for (int r = 0; r < 8; ++r) {
                int row = rbase + rt * 16 + r + 8 * hs;
                int col = cbase + ct * 16 + nloc;
                size_t idx = (size_t)row * CH + col;
                H[idx] = acc[rt][ct][r] + bias[col] + resid[idx];
            }
}

// ----------------------------------------------------- LayerNorm, 1 wave / row
__global__ __launch_bounds__(256)
void layernorm_kernel(const float* __restrict__ Hb, const float* __restrict__ gamma,
                      const float* __restrict__ beta, float* __restrict__ out) {
    int wave = threadIdx.x >> 5;
    int lane = threadIdx.x & 31;
    size_t row = (size_t)blockIdx.x * 8 + wave;
    const float* hr = Hb + row * CH;

    float s = 0.0f, ss = 0.0f;
#pragma unroll
    for (int i = 0; i < CH / 32; ++i) {
        float v = hr[lane + i * 32];
        s += v; ss += v * v;
    }
#pragma unroll
    for (int mask = 1; mask < 32; mask <<= 1) {
        s  += __shfl_xor(s,  mask, 32);
        ss += __shfl_xor(ss, mask, 32);
    }
    float mu   = s * (1.0f / CH);
    float var  = ss * (1.0f / CH) - mu * mu;
    float rstd = rsqrtf(var + 1e-5f);
#pragma unroll
    for (int i = 0; i < CH / 32; ++i) {
        int c = lane + i * 32;
        out[row * CH + c] = (hr[c] - mu) * rstd * gamma[c] + beta[c];
    }
}

extern "C" void kernel_launch(void* const* d_in, const int* in_sizes, int n_in,
                              void* d_out, int out_size, void* d_ws, size_t ws_size,
                              hipStream_t stream) {
    const float* X  = (const float*)d_in[0];
    const float* qp = (const float*)d_in[1];
    const float* Wq = (const float*)d_in[2];
    const float* bq = (const float*)d_in[3];
    const float* Wk = (const float*)d_in[4];
    const float* bk = (const float*)d_in[5];
    const float* Wv = (const float*)d_in[6];
    const float* bv = (const float*)d_in[7];
    const float* Wo = (const float*)d_in[8];
    const float* bo = (const float*)d_in[9];
    const float* gamma = (const float*)d_in[10];
    const float* beta  = (const float*)d_in[11];

    char* ws = (char*)d_ws;
    size_t off = 0;
    auto carve = [&](size_t bytes) -> void* {
        void* p = ws + off;
        off += (bytes + 255) & ~(size_t)255;
        return p;
    };
    const size_t actElems = (size_t)ROWS * CH;          // 8192*512
    const size_t wElems   = (size_t)CH * CH;            // 512*512
    _Float16* Xh  = (_Float16*)carve(actElems * 2);
    _Float16* Wqh = (_Float16*)carve(wElems * 2);
    _Float16* Wkh = (_Float16*)carve(wElems * 2);
    _Float16* Wvh = (_Float16*)carve(wElems * 2);
    _Float16* Woh = (_Float16*)carve(wElems * 2);
    _Float16* Qh  = (_Float16*)carve(actElems * 2);     // [b][h][n][d]
    _Float16* Kh  = (_Float16*)carve(actElems * 2);     // [b][h][n][d]
    _Float16* Vth = (_Float16*)carve(actElems * 2);     // [b][h][d][n]
    _Float16* Ah  = (_Float16*)carve(actElems * 2);     // merged-head attn out
    float*    Hb  = (float*)carve(actElems * 4);        // residual sum (pre-LN)

    cvt_f16_kernel<<<1024, 256, 0, stream>>>(X,  Xh,  (int)actElems);
    cvt_f16_kernel<<<256,  256, 0, stream>>>(Wq, Wqh, (int)wElems);
    cvt_f16_kernel<<<256,  256, 0, stream>>>(Wk, Wkh, (int)wElems);
    cvt_f16_kernel<<<256,  256, 0, stream>>>(Wv, Wvh, (int)wElems);
    cvt_f16_kernel<<<256,  256, 0, stream>>>(Wo, Woh, (int)wElems);

    dim3 ggrid(ROWS / 128, CH / 128);   // 64 x 4, 4 waves (128x128) per block
    gemm_qkv_kernel<<<ggrid, 128, 0, stream>>>(Xh, Wqh, bq, Qh, 0);
    gemm_qkv_kernel<<<ggrid, 128, 0, stream>>>(Xh, Wkh, bk, Kh, 0);
    gemm_qkv_kernel<<<ggrid, 128, 0, stream>>>(Xh, Wvh, bv, Vth, 1);

    flash_attn_kernel<<<dim3(SEQ / 64, BATCH * HEADS), 128, 0, stream>>>(Qh, Kh, Vth, qp, Ah);

    gemm_out_kernel<<<ggrid, 128, 0, stream>>>(Ah, Woh, bo, X, Hb);

    layernorm_kernel<<<ROWS / 8, 256, 0, stream>>>(Hb, gamma, beta, (float*)d_out);
}
